// SpatialMemoryAttention_44306882625854
// MI455X (gfx1250) — compile-verified, matrix operand order checked by reading, not run on hardware
//
#include <hip/hip_runtime.h>
#include <hip/hip_bf16.h>
#include <math.h>

// ---------------- problem constants ----------------
#define PB   32
#define PD   64
#define PH   8
#define PN   4096
#define PL   12
#define PDK  8
#define PNL  (PN * PL)           // 49152 columns per batch
#define PCOLS ((size_t)PB * PNL) // 1,572,864 total columns
#define SCALE 0.35355339059327373f // 1/sqrt(8)

typedef __attribute__((ext_vector_type(16))) _Float16 v16h;
typedef __attribute__((ext_vector_type(8)))  _Float16 v8h;
typedef __attribute__((ext_vector_type(8)))  float    v8f;

// ---------------- WMMA helper ----------------
__device__ __forceinline__ v8f wmma16(v16h a, v16h b, v8f c) {
  // D = A(16x32 f16) * B(32x16 f16) + C(16x16 f32)
  return __builtin_amdgcn_wmma_f32_16x16x32_f16(
      /*neg_a=*/false, a, /*neg_b=*/false, b,
      /*c_mod=*/(short)0, c, /*reuse_a=*/false, /*reuse_b=*/false);
}

// A fragment from a pre-converted f16 weight matrix [64x64] row-major.
// 16-bit A 16x32 layout: lanes0-15 hold M=lane, halves 0..7 = K 0..7, 8..15 = K 16..23
//                        lanes16-31 hold M=lane-16, K offset +8 (and +24..31)
// Two aligned 16B vector loads per lane, zero conversion VALU.
__device__ __forceinline__ v16h load_a_frag16(const _Float16* __restrict__ W16,
                                              int mt, int ks, int lane) {
  const int m  = 16 * mt + (lane & 15);
  const int kb = 32 * ks + ((lane & 16) ? 8 : 0);
  const _Float16* p = W16 + m * 64 + kb;    // 16B aligned
  const v8h lo = *(const v8h*)(p);
  const v8h hi = *(const v8h*)(p + 16);
  return __builtin_shufflevector(lo, hi, 0, 1, 2, 3, 4, 5, 6, 7,
                                 8, 9, 10, 11, 12, 13, 14, 15);
}

// B fragment (32x16: K rows x 16 cols) from an LDS tile laid out [64 rows][16 cols] f32
__device__ __forceinline__ v16h load_b_frag_tile(const float* t, int ks, int lane) {
  const int c  = lane & 15;
  const int kb = 32 * ks + ((lane & 16) ? 8 : 0);
  v16h b;
#pragma unroll
  for (int j = 0; j < 8; ++j) b[j]     = (_Float16)t[(kb + j) * 16 + c];
#pragma unroll
  for (int j = 0; j < 8; ++j) b[8 + j] = (_Float16)t[(kb + 16 + j) * 16 + c];
  return b;
}

// Stage one wave's 64ch x 16col f32 tile of x into LDS via the CDNA5 async
// LDS path (ASYNCcnt).  8 x (32 lanes x 16B) fully coalesced transfers.
// LDS address is the vdst VGPR plus offset; low 32 bits of a generic shared
// pointer are the LDS byte offset.
__device__ __forceinline__ void async_stage_xtile(const float* __restrict__ xb,
                                                  int colbase, float* ltile,
                                                  int lane) {
  const int sub = lane & 3;   // 16B segment within a 64B row
  const int r0  = lane >> 2;  // 0..7 : row within a group of 8 channels
  const unsigned lds0 = (unsigned)(uintptr_t)ltile;
#pragma unroll
  for (int i = 0; i < 8; ++i) {
    const int ch = i * 8 + r0;
    const unsigned long long ga =
        (unsigned long long)(xb + (size_t)ch * PNL + colbase + sub * 4);
    const unsigned la = lds0 + (unsigned)(ch * 64 + sub * 16);
    asm volatile("global_load_async_to_lds_b128 %0, %1, off"
                 :: "v"(la), "v"(ga) : "memory");
  }
  asm volatile("s_wait_asynccnt 0x0" ::: "memory");
}

// ------- KW: one-time f32 -> f16 conversion of the three 64x64 weights -------
__global__ __launch_bounds__(256) void kw_cvt(const float* __restrict__ wq,
                                              const float* __restrict__ wv,
                                              const float* __restrict__ wc,
                                              _Float16* __restrict__ w16) {
  const int i = blockIdx.x * 256 + threadIdx.x; // 0..4095
  w16[i]            = (_Float16)wq[i];
  w16[4096 + i]     = (_Float16)wv[i];
  w16[2 * 4096 + i] = (_Float16)wc[i];
}

// ---------------- K0: key_sm = softmax(memory * scale, axis=dk) ----------------
__global__ __launch_bounds__(256) void k0_keysm(const float* __restrict__ mem,
                                                float* __restrict__ keysm) {
  const int i = blockIdx.x * 256 + threadIdx.x; // flat (h,l,n), always in range
  const float* p = mem + (size_t)i * PDK;
  float v[PDK];
  float m = -1e30f;
#pragma unroll
  for (int r = 0; r < PDK; ++r) { v[r] = p[r] * SCALE; m = fmaxf(m, v[r]); }
  float s = 0.f;
#pragma unroll
  for (int r = 0; r < PDK; ++r) { v[r] = __expf(v[r] - m); s += v[r]; }
  const float inv = 1.f / s;
  float* o = keysm + (size_t)i * PDK;
#pragma unroll
  for (int r = 0; r < PDK; ++r) o[r] = v[r] * inv;
}

// ---------------- K1: v = relu(Wv x + bv); kv[b,h,l] += key_sm^T v ----------------
__global__ __launch_bounds__(256) void k1_kv(const float* __restrict__ x,
                                             const _Float16* __restrict__ wv16,
                                             const float* __restrict__ bv,
                                             const float* __restrict__ keysm,
                                             float* __restrict__ kv) {
  __shared__ float kvacc[PH * PL * PDK * PDK]; // 6144 floats = 24 KB
  __shared__ float xtile[8][PD * 16];          // 8 waves x 4 KB = 32 KB
  const int tid = threadIdx.x;
  for (int i = tid; i < PH * PL * PDK * PDK; i += 256) kvacc[i] = 0.f;
  __syncthreads();

  const int wave = tid >> 5, lane = tid & 31;
  const int tile = blockIdx.x * 8 + wave;  // one 16-column tile per wave
  const int col0 = tile * 16;
  const int b    = (int)(((size_t)col0) / PNL);
  const int cw0  = col0 - b * PNL;
  const float* xb = x + (size_t)b * PD * PNL;

  async_stage_xtile(xb, cw0, &xtile[wave][0], lane);
  const v16h bx0 = load_b_frag_tile(&xtile[wave][0], 0, lane);
  const v16h bx1 = load_b_frag_tile(&xtile[wave][0], 1, lane);

  const int myc = cw0 + (lane & 15);
  const int nn  = myc / PL;
  const int ll  = myc - nn * PL;

#pragma unroll
  for (int mt = 0; mt < 4; ++mt) {
    const int d0 = 16 * mt + ((lane & 16) ? 8 : 0);
    const int h  = d0 >> 3;
    v8f acc;
#pragma unroll
    for (int r = 0; r < 8; ++r) acc[r] = bv[d0 + r];
    acc = wmma16(load_a_frag16(wv16, mt, 0, lane), bx0, acc);
    acc = wmma16(load_a_frag16(wv16, mt, 1, lane), bx1, acc);
#pragma unroll
    for (int r = 0; r < 8; ++r) acc[r] = fmaxf(acc[r], 0.f); // v values, y = 0..7

    const float* ksp = keysm + (((size_t)h * PL + ll) * PN + nn) * PDK;
    float kx[PDK];
#pragma unroll
    for (int xi = 0; xi < PDK; ++xi) kx[xi] = ksp[xi];

    float* base = kvacc + (h * PL + ll) * 64;
#pragma unroll
    for (int xi = 0; xi < PDK; ++xi)
#pragma unroll
      for (int y = 0; y < PDK; ++y)
        atomicAdd(&base[xi * PDK + y], kx[xi] * acc[y]); // ds_add_f32
  }
  __syncthreads();

  const int bb = (int)(((size_t)blockIdx.x * 128) / PNL);
  float* kvb = kv + (size_t)bb * (PH * PL * PDK * PDK);
  for (int i = tid; i < PH * PL * PDK * PDK; i += 256)
    atomicAdd(&kvb[i], kvacc[i]);    // global_atomic_add_f32
}

// ---------------- K2: full fused output ----------------
__global__ __launch_bounds__(256) void k2_out(const float* __restrict__ x,
                                              const _Float16* __restrict__ w16,
                                              const float* __restrict__ bq,
                                              const float* __restrict__ bv,
                                              const float* __restrict__ bc,
                                              const float* __restrict__ kv,
                                              const float* __restrict__ weight,
                                              const float* __restrict__ bias,
                                              float* __restrict__ out) {
  // One 4KB tile per wave, reused: first the staged x tile, then the y tile.
  // Same-wave DS ops complete in order, and the async stage is fenced by
  // s_wait_asynccnt, so no block barrier is needed (waves are independent).
  __shared__ float tb[8][PD * 16]; // 32 KB
  const int tid = threadIdx.x, wave = tid >> 5, lane = tid & 31;
  const int tile = blockIdx.x * 8 + wave;
  const int col0 = tile * 16;
  const int b    = (int)(((size_t)col0) / PNL);
  const int cw0  = col0 - b * PNL;
  const float* xb = x + (size_t)b * PD * PNL;
  float* t = &tb[wave][0];

  const _Float16* wq16 = w16;
  const _Float16* wv16 = w16 + 4096;
  const _Float16* wc16 = w16 + 2 * 4096;

  __builtin_prefetch(weight + cw0, 0, 3);   // global_prefetch_b8
  __builtin_prefetch(bias + cw0, 0, 3);

  async_stage_xtile(xb, cw0, t, lane);
  const v16h bx0 = load_b_frag_tile(t, 0, lane);
  const v16h bx1 = load_b_frag_tile(t, 1, lane);

  const int myc = cw0 + (lane & 15);
  const int nn  = myc / PL;
  const int ll  = myc - nn * PL;

#pragma unroll
  for (int mt = 0; mt < 4; ++mt) {
    const int d0 = 16 * mt + ((lane & 16) ? 8 : 0);
    const int h  = d0 >> 3;
    v8f aq, av;
#pragma unroll
    for (int r = 0; r < 8; ++r) { aq[r] = bq[d0 + r]; av[r] = bv[d0 + r]; }
    aq = wmma16(load_a_frag16(wq16, mt, 0, lane), bx0, aq);
    aq = wmma16(load_a_frag16(wq16, mt, 1, lane), bx1, aq);
    av = wmma16(load_a_frag16(wv16, mt, 0, lane), bx0, av);
    av = wmma16(load_a_frag16(wv16, mt, 1, lane), bx1, av);
#pragma unroll
    for (int r = 0; r < 8; ++r) { aq[r] = fmaxf(aq[r], 0.f); av[r] = fmaxf(av[r], 0.f); }

    // per-lane softmax over dk (this lane's 8 acc regs are exactly head h's dk vals)
    float m = aq[0];
#pragma unroll
    for (int r = 1; r < 8; ++r) m = fmaxf(m, aq[r]);
    float e[8], s = 0.f;
#pragma unroll
    for (int r = 0; r < 8; ++r) { e[r] = __expf((aq[r] - m) * SCALE); s += e[r]; }
    const float inv = 1.f / s;

    // y = q_sm @ kv + v   (row-sum of Aapt softmax == 1  =>  attn_dyn == v5)
    const float* kvp = kv + (((size_t)b * PH + h) * PL + ll) * 64;
#pragma unroll
    for (int y = 0; y < PDK; ++y) {
      float acc = av[y];
#pragma unroll
      for (int xi = 0; xi < PDK; ++xi) acc += (e[xi] * inv) * kvp[xi * PDK + y];
      t[(d0 + y) * 16 + (lane & 15)] = acc; // overwrite x tile with y tile
    }
  }

  const v16h by0 = load_b_frag_tile(t, 0, lane);
  const v16h by1 = load_b_frag_tile(t, 1, lane);

#pragma unroll
  for (int mt = 0; mt < 4; ++mt) {
    const int d0 = 16 * mt + ((lane & 16) ? 8 : 0);
    v8f ac;
#pragma unroll
    for (int r = 0; r < 8; ++r) ac[r] = bc[d0 + r];
    ac = wmma16(load_a_frag16(wc16, mt, 0, lane), by0, ac);
    ac = wmma16(load_a_frag16(wc16, mt, 1, lane), by1, ac);
#pragma unroll
    for (int r = 0; r < 8; ++r) {
      const float z = fmaxf(ac[r], 0.f);
      const size_t wb = (size_t)(d0 + r) * PNL + myc;
      out[(size_t)b * PD * PNL + wb] = z * weight[wb] + bias[wb] + z;
    }
  }
}

// ---------------- host launch ----------------
extern "C" void kernel_launch(void* const* d_in, const int* in_sizes, int n_in,
                              void* d_out, int out_size, void* d_ws, size_t ws_size,
                              hipStream_t stream) {
  const float* x      = (const float*)d_in[0];
  const float* wq     = (const float*)d_in[1];
  const float* bq     = (const float*)d_in[2];
  const float* wv     = (const float*)d_in[3];
  const float* bv     = (const float*)d_in[4];
  const float* wc     = (const float*)d_in[5];
  const float* bc     = (const float*)d_in[6];
  const float* mem    = (const float*)d_in[7];
  const float* weight = (const float*)d_in[8];
  const float* bias   = (const float*)d_in[9];
  // d_in[10], d_in[11] (nv1, nv2) are mathematically dead: row-sums of the
  // softmax are identically 1, so attn_dyn == v5.
  float* out = (float*)d_out;

  // workspace layout (floats): keysm | kv | w16 (as halves)
  float* keysm = (float*)d_ws;                         // H*L*N*DK = 12.6 MB
  float* kv    = keysm + (size_t)PH * PL * PN * PDK;   // B*H*L*64 = 0.79 MB
  _Float16* w16 = (_Float16*)(kv + (size_t)PB * PH * PL * PDK * PDK); // 3*4096 halves

  hipMemsetAsync(kv, 0, sizeof(float) * (size_t)PB * PH * PL * PDK * PDK, stream);

  kw_cvt<<<16, 256, 0, stream>>>(wq, wv, wc, w16);
  k0_keysm<<<(PH * PL * PN) / 256, 256, 0, stream>>>(mem, keysm);

  const int nblocks = (int)(PCOLS / 128); // 8 waves/block, 16 cols/wave
  k1_kv<<<nblocks, 256, 0, stream>>>(x, w16 + 4096, bv, keysm, kv);
  k2_out<<<nblocks, 256, 0, stream>>>(x, w16, bq, bv, bc, kv,
                                      weight, bias, out);
}
